// Agent1_78675210928757
// MI455X (gfx1250) — matrix operand order repaired and checked by reading, not compiled, not run on hardware
//
#include <hip/hip_runtime.h>

// ---------------------------------------------------------------------------
// Types for CDNA5 WMMA
// ---------------------------------------------------------------------------
typedef __bf16 v16bf __attribute__((ext_vector_type(16)));
typedef float  v8f   __attribute__((ext_vector_type(8)));

union FragBF { uint4 q[2]; v16bf v; unsigned short s[16]; };

// Branchless round-to-nearest-even f32 -> bf16.
__device__ __forceinline__ unsigned short f2bf(float f) {
    unsigned int u = __float_as_uint(f);
    u += 0x7fffu + ((u >> 16) & 1u);
    return (unsigned short)(u >> 16);
}

__device__ __forceinline__ float nan0f(float x) { return (x != x) ? 0.0f : x; }

// CDNA5 async global->LDS copy (per-lane, ASYNCcnt-tracked, no VGPR round trip)
__device__ __forceinline__ void async_ld_lds_b128(unsigned lds_addr,
                                                  const void* gptr) {
    unsigned long long ga = (unsigned long long)(size_t)gptr;
    asm volatile("global_load_async_to_lds_b128 %0, %1, off"
                 :: "v"(lds_addr), "v"(ga) : "memory");
}
__device__ __forceinline__ void wait_async() {
    asm volatile("s_wait_asynccnt 0x0" ::: "memory");
}

#define EPS 1e-5f

// ---------------------------------------------------------------------------
// f32 -> bf16 array conversion (weight pre-pass)
// ---------------------------------------------------------------------------
__global__ void k_f2bf_arr(const float* __restrict__ s,
                           unsigned short* __restrict__ d, int n) {
    const int i = blockIdx.x * 256 + threadIdx.x;
    if (i < n) d[i] = f2bf(s[i]);
}

// ---------------------------------------------------------------------------
// WMMA GEMM:  C[M,N] = (A_bf16[M,K] @ W_bf16[N,K]^T + bias) * oscale
// block = 128 threads (4 waves); 64x64 block tile; each wave owns a 32x32
// quadrant computed as 2x2 WMMA tiles -> 4 v_wmma per wave per k-step.
// Double-buffered LDS staging via global_load_async_to_lds_b128: the DMA for
// tile k+1 is in flight while the WMMAs for tile k execute.
// ---------------------------------------------------------------------------
__global__ void k_gemm_bf16(const unsigned short* __restrict__ A,
                            const unsigned short* __restrict__ W,
                            const float* __restrict__ bias,
                            float* __restrict__ Cf,
                            unsigned short* __restrict__ Cb,
                            int M, int N, int K, float oscale) {
    const int m0   = blockIdx.x * 64;
    const int n0   = blockIdx.y * 64;
    const int tid  = threadIdx.x;
    const int wave = tid >> 5;
    const int lane = tid & 31;
    const int mi2  = (wave >> 1) * 32;     // quadrant row offset
    const int ni2  = (wave & 1) * 32;      // quadrant col offset
    const int l16  = lane & 15, hi = lane >> 4;
    const int kb   = hi * 8;

    __shared__ __align__(16) unsigned short As[2][64][40];
    __shared__ __align__(16) unsigned short Bs[2][64][40];

    v8f c00 = {}, c01 = {}, c10 = {}, c11 = {};

    const int lr   = tid >> 1;             // 0..63: staged row
    const int lseg = (tid & 1) * 16;       // 0 or 16 (halves)
    const unsigned ldsA[2] = { (unsigned)(size_t)&As[0][lr][lseg],
                               (unsigned)(size_t)&As[1][lr][lseg] };
    const unsigned ldsB[2] = { (unsigned)(size_t)&Bs[0][lr][lseg],
                               (unsigned)(size_t)&Bs[1][lr][lseg] };

    const unsigned short* ap = A + (size_t)(m0 + lr) * K + lseg;
    const unsigned short* wp = W + (size_t)(n0 + lr) * K + lseg;

    // prologue: DMA tile 0 into buffer 0
    async_ld_lds_b128(ldsA[0],      ap);
    async_ld_lds_b128(ldsA[0] + 16, ap + 8);
    async_ld_lds_b128(ldsB[0],      wp);
    async_ld_lds_b128(ldsB[0] + 16, wp + 8);

    for (int k0 = 0; k0 < K; k0 += 32) {
        const int cur = (k0 >> 5) & 1;
        wait_async();      // tile issued last iteration has landed
        __syncthreads();   // visible to all waves; prior reads of nxt done

        if (k0 + 32 < K) { // fire-and-forget DMA for next tile
            const int nxt = cur ^ 1;
            const unsigned short* apn = ap + k0 + 32;
            const unsigned short* wpn = wp + k0 + 32;
            async_ld_lds_b128(ldsA[nxt],      apn);
            async_ld_lds_b128(ldsA[nxt] + 16, apn + 8);
            async_ld_lds_b128(ldsB[nxt],      wpn);
            async_ld_lds_b128(ldsB[nxt] + 16, wpn + 8);
        }

        FragBF a0, a1, b0, b1;
        a0.q[0] = *(const uint4*)&As[cur][mi2 + l16][kb];
        a0.q[1] = *(const uint4*)&As[cur][mi2 + l16][16 + kb];
        a1.q[0] = *(const uint4*)&As[cur][mi2 + 16 + l16][kb];
        a1.q[1] = *(const uint4*)&As[cur][mi2 + 16 + l16][16 + kb];
        b0.q[0] = *(const uint4*)&Bs[cur][ni2 + l16][kb];
        b0.q[1] = *(const uint4*)&Bs[cur][ni2 + l16][16 + kb];
        b1.q[0] = *(const uint4*)&Bs[cur][ni2 + 16 + l16][kb];
        b1.q[1] = *(const uint4*)&Bs[cur][ni2 + 16 + l16][16 + kb];

        c00 = __builtin_amdgcn_wmma_f32_16x16x32_bf16(
                  false, a0.v, false, b0.v, (short)0, c00, false, false);
        c01 = __builtin_amdgcn_wmma_f32_16x16x32_bf16(
                  false, a0.v, false, b1.v, (short)0, c01, false, false);
        c10 = __builtin_amdgcn_wmma_f32_16x16x32_bf16(
                  false, a1.v, false, b0.v, (short)0, c10, false, false);
        c11 = __builtin_amdgcn_wmma_f32_16x16x32_bf16(
                  false, a1.v, false, b1.v, (short)0, c11, false, false);
    }

    const int nc0 = n0 + ni2 + l16;
    const int nc1 = nc0 + 16;
    const float bv0 = bias ? bias[nc0] : 0.0f;
    const float bv1 = bias ? bias[nc1] : 0.0f;
#pragma unroll
    for (int r = 0; r < 8; ++r) {
        const int row0 = m0 + mi2 + hi * 8 + r;
        const int row1 = row0 + 16;
        const float v00 = (c00[r] + bv0) * oscale;
        const float v01 = (c01[r] + bv1) * oscale;
        const float v10 = (c10[r] + bv0) * oscale;
        const float v11 = (c11[r] + bv1) * oscale;
        if (Cf) {
            Cf[(size_t)row0 * N + nc0] = v00;
            Cf[(size_t)row0 * N + nc1] = v01;
            Cf[(size_t)row1 * N + nc0] = v10;
            Cf[(size_t)row1 * N + nc1] = v11;
        }
        if (Cb) {
            Cb[(size_t)row0 * N + nc0] = f2bf(v00);
            Cb[(size_t)row0 * N + nc1] = f2bf(v01);
            Cb[(size_t)row1 * N + nc0] = f2bf(v10);
            Cb[(size_t)row1 * N + nc1] = f2bf(v11);
        }
    }
}

// ---------------------------------------------------------------------------
// Flash attention (per head, 64 query rows per block, key chunks of 32).
// block = 128 threads (4 waves, 16 q-rows per wave).
// Double-buffered K (async DMA) and V (load + LDS transpose) staging.
// ---------------------------------------------------------------------------
__global__ void k_flash(const unsigned short* __restrict__ Q,
                        const unsigned short* __restrict__ Kb,
                        const unsigned short* __restrict__ Vb,
                        unsigned short* __restrict__ Ctx, int N) {
    const int head = blockIdx.y;
    const int q0   = blockIdx.x * 64;
    const int tid  = threadIdx.x;
    const int wave = tid >> 5;
    const int lane = tid & 31;
    const int l16  = lane & 15, hi = lane >> 4;
    const int kb   = hi * 8;

    __shared__ __align__(16) unsigned short Ks[2][32][40];
    __shared__ __align__(16) unsigned short Vt[2][32][40];  // [d][key]
    __shared__ __align__(16) float          Scr[4][16][33];
    __shared__ __align__(16) unsigned short Pl[4][16][40];
    __shared__ float Ml[4][16], Ll[4][16], Al[4][16];

    const int qrow = q0 + wave * 16 + l16;
    const unsigned short* qp = Q + (size_t)qrow * 192 + head * 32;
    FragBF qa;
    qa.q[0] = *(const uint4*)(qp + kb);
    qa.q[1] = *(const uint4*)(qp + 16 + kb);

    const int sr   = tid >> 2;            // staged row 0..31
    const int sseg = (tid & 3) * 8;       // halves 0,8,16,24
    const unsigned ldsK[2] = { (unsigned)(size_t)&Ks[0][sr][sseg],
                               (unsigned)(size_t)&Ks[1][sr][sseg] };
    const unsigned short* kpb = Kb + (size_t)sr * 192 + head * 32 + sseg;
    const unsigned short* vpb = Vb + (size_t)sr * 192 + head * 32 + sseg;

    v8f o0 = {}, o1 = {};
    if (lane < 16) { Ml[wave][lane] = -1e30f; Ll[wave][lane] = 0.0f; }

    // prologue: stage key chunk 0 into buffer 0
    async_ld_lds_b128(ldsK[0], kpb);
    {
        uint4 vv = *(const uint4*)vpb;
        const unsigned short* vs = (const unsigned short*)&vv;
#pragma unroll
        for (int j = 0; j < 8; ++j) Vt[0][sseg + j][sr] = vs[j];
    }

    for (int kc = 0; kc < N; kc += 32) {
        const int cur = (kc >> 5) & 1;
        wait_async();      // previous iteration's K DMA landed
        __syncthreads();   // all writes visible; prior reads of nxt done

        if (kc + 32 < N) { // stage next chunk into the other buffer
            const int nxt = cur ^ 1;
            async_ld_lds_b128(ldsK[nxt], kpb + (size_t)(kc + 32) * 192);
            uint4 vv = *(const uint4*)(vpb + (size_t)(kc + 32) * 192);
            const unsigned short* vs = (const unsigned short*)&vv;
#pragma unroll
            for (int j = 0; j < 8; ++j) Vt[nxt][sseg + j][sr] = vs[j];
        }

        // S = Q K^T : two 16x16 tiles (keys 0-15, 16-31)
        FragBF kf0, kf1;
        kf0.q[0] = *(const uint4*)&Ks[cur][l16][kb];
        kf0.q[1] = *(const uint4*)&Ks[cur][l16][16 + kb];
        kf1.q[0] = *(const uint4*)&Ks[cur][16 + l16][kb];
        kf1.q[1] = *(const uint4*)&Ks[cur][16 + l16][16 + kb];
        v8f s0 = {}, s1 = {};
        s0 = __builtin_amdgcn_wmma_f32_16x16x32_bf16(
                 false, qa.v, false, kf0.v, (short)0, s0, false, false);
        s1 = __builtin_amdgcn_wmma_f32_16x16x32_bf16(
                 false, qa.v, false, kf1.v, (short)0, s1, false, false);

#pragma unroll
        for (int r = 0; r < 8; ++r) {
            Scr[wave][hi * 8 + r][l16]      = s0[r];
            Scr[wave][hi * 8 + r][16 + l16] = s1[r];
        }
        asm volatile("s_wait_dscnt 0x0" ::: "memory");

        // online softmax: lanes 0..15 each own one q row
        if (lane < 16) {
            float mold = Ml[wave][lane];
            float mx = mold;
#pragma unroll
            for (int j = 0; j < 32; ++j) mx = fmaxf(mx, Scr[wave][lane][j]);
            float alpha = __expf(mold - mx);
            float lsum  = Ll[wave][lane] * alpha;
#pragma unroll
            for (int j = 0; j < 32; ++j) {
                float p = __expf(Scr[wave][lane][j] - mx);
                lsum += p;
                Pl[wave][lane][j] = f2bf(p);
            }
            Ml[wave][lane] = mx; Ll[wave][lane] = lsum; Al[wave][lane] = alpha;
        }
        asm volatile("s_wait_dscnt 0x0" ::: "memory");

#pragma unroll
        for (int r = 0; r < 8; ++r) {
            float a = Al[wave][hi * 8 + r];
            o0[r] *= a; o1[r] *= a;
        }

        FragBF pf, vf0, vf1;
        pf.q[0]  = *(const uint4*)&Pl[wave][l16][kb];
        pf.q[1]  = *(const uint4*)&Pl[wave][l16][16 + kb];
        vf0.q[0] = *(const uint4*)&Vt[cur][l16][kb];
        vf0.q[1] = *(const uint4*)&Vt[cur][l16][16 + kb];
        vf1.q[0] = *(const uint4*)&Vt[cur][16 + l16][kb];
        vf1.q[1] = *(const uint4*)&Vt[cur][16 + l16][16 + kb];
        o0 = __builtin_amdgcn_wmma_f32_16x16x32_bf16(
                 false, pf.v, false, vf0.v, (short)0, o0, false, false);
        o1 = __builtin_amdgcn_wmma_f32_16x16x32_bf16(
                 false, pf.v, false, vf1.v, (short)0, o1, false, false);
    }

    asm volatile("s_wait_dscnt 0x0" ::: "memory");
#pragma unroll
    for (int r = 0; r < 8; ++r) {
        const int row = q0 + wave * 16 + hi * 8 + r;
        const float inv = 1.0f / Ll[wave][hi * 8 + r];
        unsigned short* cp = Ctx + (size_t)row * 192 + head * 32;
        cp[l16]      = f2bf(nan0f(o0[r] * inv));   // reference: _nan0(ctx)
        cp[16 + l16] = f2bf(nan0f(o1[r] * inv));
    }
}

// ---------------------------------------------------------------------------
// Row LayerNorm over 256 columns, optional residual, optional bf16 mirror.
// ---------------------------------------------------------------------------
__global__ void k_ln_rows(const float* __restrict__ X,
                          const float* __restrict__ R,
                          const float* __restrict__ g,
                          const float* __restrict__ b,
                          float* __restrict__ Of,
                          unsigned short* __restrict__ Ob) {
    const int row = blockIdx.x, c = threadIdx.x;
    __shared__ float red[256];
    float v = X[(size_t)row * 256 + c];
    if (R) v += R[(size_t)row * 256 + c];
    red[c] = v; __syncthreads();
    for (int s = 128; s > 0; s >>= 1) {
        if (c < s) red[c] += red[c + s];
        __syncthreads();
    }
    const float mean = red[0] * (1.0f / 256.0f);
    __syncthreads();
    const float d = v - mean;
    red[c] = d * d; __syncthreads();
    for (int s = 128; s > 0; s >>= 1) {
        if (c < s) red[c] += red[c + s];
        __syncthreads();
    }
    const float var = red[0] * (1.0f / 256.0f);
    const float y = d * rsqrtf(var + EPS) * g[c] + b[c];
    if (Of) Of[(size_t)row * 256 + c] = y;
    if (Ob) Ob[(size_t)row * 256 + c] = f2bf(y);
}

// ---------------------------------------------------------------------------
// Deterministic BatchNorm column statistics (two stage).
// ---------------------------------------------------------------------------
__global__ void k_bn_partial(const float* __restrict__ X, float* __restrict__ part) {
    const int blk = blockIdx.x;   // 16 blocks of 256 rows
    const int c   = threadIdx.x;  // column
    float s = 0.0f, ss = 0.0f;
    for (int r = 0; r < 256; ++r) {
        float v = X[(size_t)(blk * 256 + r) * 256 + c];
        s += v; ss += v * v;
    }
    part[((blk << 8) + c) * 2 + 0] = s;
    part[((blk << 8) + c) * 2 + 1] = ss;
}

__global__ void k_bn_final(const float* __restrict__ part, float* __restrict__ st) {
    const int c = threadIdx.x;
    float s = 0.0f, ss = 0.0f;
    for (int b = 0; b < 16; ++b) {
        s  += part[((b << 8) + c) * 2 + 0];
        ss += part[((b << 8) + c) * 2 + 1];
    }
    const float mean = s * (1.0f / 4096.0f);
    const float var  = ss * (1.0f / 4096.0f) - mean * mean;
    st[2 * c]     = mean;
    st[2 * c + 1] = rsqrtf(var + EPS);
}

__global__ void k_bn_apply_bf16(const float* __restrict__ X,
                                const float* __restrict__ st,
                                const float* __restrict__ g,
                                const float* __restrict__ b,
                                unsigned short* __restrict__ O) {
    const int i = blockIdx.x * 256 + threadIdx.x;
    const int c = i & 255;
    const float v = (X[i] - st[2 * c]) * st[2 * c + 1] * g[c] + b[c];
    O[i] = f2bf(v);
}

// BN2 + nan0 + pos_emb add, write f32 and bf16 mirrors.
__global__ void k_pos(const float* __restrict__ H,
                      const float* __restrict__ st,
                      const float* __restrict__ g,
                      const float* __restrict__ b,
                      const float* __restrict__ pe,
                      float* __restrict__ Pf,
                      unsigned short* __restrict__ Pb) {
    const int i = blockIdx.x * 256 + threadIdx.x;
    const int c = i & 255;
    float v = (H[i] - st[2 * c]) * st[2 * c + 1] * g[c] + b[c];
    v = nan0f(v) + pe[i];
    Pf[i] = v;
    Pb[i] = f2bf(v);
}

// ---------------------------------------------------------------------------
// Segment mean over 64 groups of 64 rows (with nan0 on input).
// ---------------------------------------------------------------------------
__global__ void k_segmean(const float* __restrict__ E,
                          float* __restrict__ em_ws,
                          float* __restrict__ em_out) {
    const int grp = blockIdx.x, c = threadIdx.x;
    float s = 0.0f;
    for (int i = 0; i < 64; ++i)
        s += nan0f(E[(size_t)(grp * 64 + i) * 256 + c]);
    const float m = s * (1.0f / 64.0f);
    em_ws[grp * 256 + c]  = m;
    em_out[grp * 256 + c] = m;
}

// ---------------------------------------------------------------------------
// Head MLP: relu(enc_mean @ fc1^T + b) @ out_w^T + b, weight set by for_next.
// ---------------------------------------------------------------------------
__global__ void k_mlp(const float* __restrict__ EM,
                      const float* __restrict__ f1w_e, const float* __restrict__ f1b_e,
                      const float* __restrict__ ow_e,  const float* __restrict__ ob_e,
                      const float* __restrict__ f1w_t, const float* __restrict__ f1b_t,
                      const float* __restrict__ ow_t,  const float* __restrict__ ob_t,
                      const int* __restrict__ for_next,
                      float* __restrict__ out) {
    const int grp = blockIdx.x, t = threadIdx.x;   // 128 threads
    const int fn = *for_next;
    const float* fw = fn ? f1w_t : f1w_e;
    const float* fb = fn ? f1b_t : f1b_e;
    const float* ow = fn ? ow_t  : ow_e;
    const float* ob = fn ? ob_t  : ob_e;

    __shared__ float hid[128];
    const float* xr = EM + grp * 256;
    float acc = fb[t];
    for (int k = 0; k < 256; ++k) acc += xr[k] * fw[t * 256 + k];
    hid[t] = fmaxf(acc, 0.0f);
    __syncthreads();
    if (t < 15) {
        float a = ob[t];
        for (int k = 0; k < 128; ++k) a += hid[k] * ow[t * 128 + k];
        out[grp * 15 + t] = nan0f(a);
    }
}

// ---------------------------------------------------------------------------
// Workspace layout (bytes, 16B aligned)
// ---------------------------------------------------------------------------
#define OFF_XN     (size_t)0            // 4 MiB  f32 [4096,256]
#define OFF_HRAW   (OFF_XN   + 4194304)
#define OFF_POSF   (OFF_HRAW + 4194304)
#define OFF_ENCF   (OFF_POSF + 4194304)
#define OFF_TMPF   (OFF_ENCF + 4194304) // attn_out then w1_out
#define OFF_EMBF   (OFF_TMPF + 4194304)
#define OFF_ABF    (OFF_EMBF + 4194304) // 2 MiB bf16 [4096,256]
#define OFF_POSB   (OFF_ABF  + 2097152)
#define OFF_ENCB   (OFF_POSB + 2097152)
#define OFF_QB     (OFF_ENCB + 2097152) // 1.5 MiB bf16 [4096,192]
#define OFF_KB     (OFF_QB   + 1572864)
#define OFF_VB     (OFF_KB   + 1572864)
#define OFF_CTXB   (OFF_VB   + 1572864)
#define OFF_PART   (OFF_CTXB + 1572864) // 32 KiB
#define OFF_ST1    (OFF_PART + 32768)
#define OFF_ST2    (OFF_ST1  + 2048)
#define OFF_EMEAN  (OFF_ST2  + 2048)    // 64 KiB
#define OFF_WDL    (OFF_EMEAN + 65536)  // bf16 weights below
#define OFF_WQ     (OFF_WDL  + 131072)  // 192x256
#define OFF_WK     (OFF_WQ   + 98304)   // 192x256
#define OFF_WV     (OFF_WK   + 98304)
#define OFF_W3     (OFF_WV   + 98304)   // 256x192
#define OFF_W1     (OFF_W3   + 98304)   // 256x256

extern "C" void kernel_launch(void* const* d_in, const int* in_sizes, int n_in,
                              void* d_out, int out_size, void* d_ws, size_t ws_size,
                              hipStream_t stream) {
    (void)in_sizes; (void)n_in; (void)out_size; (void)ws_size;
    const float* x      = (const float*)d_in[0];
    const float* pe     = (const float*)d_in[1];
    const float* ln_w   = (const float*)d_in[2];
    const float* ln_b   = (const float*)d_in[3];
    const float* bn1_g  = (const float*)d_in[4];
    const float* bn1_b  = (const float*)d_in[5];
    const float* dl_w   = (const float*)d_in[6];
    const float* dl_b   = (const float*)d_in[7];
    const float* bn2_g  = (const float*)d_in[8];
    const float* bn2_b  = (const float*)d_in[9];
    const float* wq     = (const float*)d_in[10];
    const float* bq     = (const float*)d_in[11];
    const float* wk     = (const float*)d_in[12];
    const float* bk     = (const float*)d_in[13];
    const float* wv     = (const float*)d_in[14];
    const float* bv     = (const float*)d_in[15];
    const float* w3     = (const float*)d_in[16];
    const float* b3     = (const float*)d_in[17];
    const float* aln_w  = (const float*)d_in[18];
    const float* aln_b  = (const float*)d_in[19];
    const float* w1     = (const float*)d_in[20];
    const float* b1     = (const float*)d_in[21];
    const float* f1w_e  = (const float*)d_in[22];
    const float* f1b_e  = (const float*)d_in[23];
    const float* ow_e   = (const float*)d_in[24];
    const float* ob_e   = (const float*)d_in[25];
    const float* f1w_t  = (const float*)d_in[26];
    const float* f1b_t  = (const float*)d_in[27];
    const float* ow_t   = (const float*)d_in[28];
    const float* ob_t   = (const float*)d_in[29];
    const int*   fnx    = (const int*)d_in[30];

    char* ws = (char*)d_ws;
    float* xn    = (float*)(ws + OFF_XN);
    float* hraw  = (float*)(ws + OFF_HRAW);
    float* posf  = (float*)(ws + OFF_POSF);
    float* encf  = (float*)(ws + OFF_ENCF);
    float* tmpf  = (float*)(ws + OFF_TMPF);
    float* embf  = (float*)(ws + OFF_EMBF);
    unsigned short* abf  = (unsigned short*)(ws + OFF_ABF);
    unsigned short* posb = (unsigned short*)(ws + OFF_POSB);
    unsigned short* encb = (unsigned short*)(ws + OFF_ENCB);
    unsigned short* qb   = (unsigned short*)(ws + OFF_QB);
    unsigned short* kbuf = (unsigned short*)(ws + OFF_KB);
    unsigned short* vbuf = (unsigned short*)(ws + OFF_VB);
    unsigned short* ctxb = (unsigned short*)(ws + OFF_CTXB);
    float* part  = (float*)(ws + OFF_PART);
    float* st1   = (float*)(ws + OFF_ST1);
    float* st2   = (float*)(ws + OFF_ST2);
    float* emean = (float*)(ws + OFF_EMEAN);
    unsigned short* wdl_b16 = (unsigned short*)(ws + OFF_WDL);
    unsigned short* wq_b16  = (unsigned short*)(ws + OFF_WQ);
    unsigned short* wk_b16  = (unsigned short*)(ws + OFF_WK);
    unsigned short* wv_b16  = (unsigned short*)(ws + OFF_WV);
    unsigned short* w3_b16  = (unsigned short*)(ws + OFF_W3);
    unsigned short* w1_b16  = (unsigned short*)(ws + OFF_W1);
    float* out   = (float*)d_out;

    const float qscale = 0.17677669529663687f;  // 1/sqrt(32)

    // 0) pre-convert all GEMM weights to bf16
    k_f2bf_arr<<<256, 256, 0, stream>>>(dl_w, wdl_b16, 65536);
    k_f2bf_arr<<<192, 256, 0, stream>>>(wq, wq_b16, 49152);
    k_f2bf_arr<<<192, 256, 0, stream>>>(wk, wk_b16, 49152);
    k_f2bf_arr<<<192, 256, 0, stream>>>(wv, wv_b16, 49152);
    k_f2bf_arr<<<192, 256, 0, stream>>>(w3, w3_b16, 49152);
    k_f2bf_arr<<<256, 256, 0, stream>>>(w1, w1_b16, 65536);

    // 1) LayerNorm rows of x
    k_ln_rows<<<4096, 256, 0, stream>>>(x, nullptr, ln_w, ln_b, xn, nullptr);
    // 2) BN1 stats + apply -> bf16 A
    k_bn_partial<<<16, 256, 0, stream>>>(xn, part);
    k_bn_final<<<1, 256, 0, stream>>>(part, st1);
    k_bn_apply_bf16<<<4096, 256, 0, stream>>>(xn, st1, bn1_g, bn1_b, abf);
    // 3) h = A @ dl_w^T + dl_b
    k_gemm_bf16<<<dim3(64, 4), 128, 0, stream>>>(abf, wdl_b16, dl_b, hraw, nullptr,
                                                 4096, 256, 256, 1.0f);
    // 4) BN2 stats + nan0 + pos_emb
    k_bn_partial<<<16, 256, 0, stream>>>(hraw, part);
    k_bn_final<<<1, 256, 0, stream>>>(part, st2);
    k_pos<<<4096, 256, 0, stream>>>(hraw, st2, bn2_g, bn2_b, pe, posf, posb);
    // 5) Q/K/V projections (1/sqrt(32) folded into Q)
    k_gemm_bf16<<<dim3(64, 3), 128, 0, stream>>>(posb, wq_b16, bq, nullptr, qb,
                                                 4096, 192, 256, qscale);
    k_gemm_bf16<<<dim3(64, 3), 128, 0, stream>>>(posb, wk_b16, bk, nullptr, kbuf,
                                                 4096, 192, 256, 1.0f);
    k_gemm_bf16<<<dim3(64, 3), 128, 0, stream>>>(posb, wv_b16, bv, nullptr, vbuf,
                                                 4096, 192, 256, 1.0f);
    // 6) flash attention -> ctx bf16 [4096,192]
    k_flash<<<dim3(64, 6), 128, 0, stream>>>(qb, kbuf, vbuf, ctxb, 4096);
    // 7) ctx @ w3^T + b3 ; enc = LN(pos + out)
    k_gemm_bf16<<<dim3(64, 4), 128, 0, stream>>>(ctxb, w3_b16, b3, tmpf, nullptr,
                                                 4096, 256, 192, 1.0f);
    k_ln_rows<<<4096, 256, 0, stream>>>(tmpf, posf, aln_w, aln_b, encf, encb);
    // 8) enc @ w1^T + b1 ; emb = LN(enc + out)
    k_gemm_bf16<<<dim3(64, 4), 128, 0, stream>>>(encb, w1_b16, b1, tmpf, nullptr,
                                                 4096, 256, 256, 1.0f);
    k_ln_rows<<<4096, 256, 0, stream>>>(tmpf, encf, aln_w, aln_b, embf, nullptr);
    // 9) segment mean -> d_out[960..], ws copy for MLP
    k_segmean<<<64, 256, 0, stream>>>(embf, emean, out + 960);
    // 10) head MLP -> logits d_out[0..959]
    k_mlp<<<64, 128, 0, stream>>>(emean, f1w_e, f1b_e, ow_e, ob_e,
                                  f1w_t, f1b_t, ow_t, ob_t, fnx, out);
}
